// Head_33767032881197
// MI455X (gfx1250) — matrix operand order repaired and checked by reading, not compiled, not run on hardware
//
#include <hip/hip_runtime.h>
#include <stdint.h>

#define B_ 8
#define T_ 2048
#define C_ 1024
#define H_ 128

typedef __attribute__((ext_vector_type(16))) __bf16 v16bf;
typedef __attribute__((ext_vector_type(8)))  float  v8f;

#if defined(__has_builtin)
#if __has_builtin(__builtin_amdgcn_cvt_pk_bf16_f32)
#define HAVE_CVT_PK_BF16 1
#endif
#endif

union Frag16 {
  uint32_t u[8];
  uint4    q[2];
  v16bf    b;
};

__device__ __forceinline__ uint32_t pack2_bf16(float lo, float hi) {
#ifdef HAVE_CVT_PK_BF16
  // single v_cvt_pk_bf16_f32 (RNE), replaces 5-VALU manual sequence
  auto r = __builtin_amdgcn_cvt_pk_bf16_f32(lo, hi);
  return __builtin_bit_cast(uint32_t, r);
#else
  uint32_t a = __builtin_bit_cast(uint32_t, lo);
  uint32_t c = __builtin_bit_cast(uint32_t, hi);
  a += 0x7FFFu + ((a >> 16) & 1u);   // round-to-nearest-even bf16
  c += 0x7FFFu + ((c >> 16) & 1u);
  return (a >> 16) | (c & 0xFFFF0000u);
#endif
}
__device__ __forceinline__ uint16_t f2bf(float f) {
#ifdef HAVE_CVT_PK_BF16
  auto r = __builtin_amdgcn_cvt_pk_bf16_f32(f, f);
  return (uint16_t)(__builtin_bit_cast(uint32_t, r) & 0xFFFFu);
#else
  uint32_t a = __builtin_bit_cast(uint32_t, f);
  a += 0x7FFFu + ((a >> 16) & 1u);
  return (uint16_t)(a >> 16);
#endif
}

// ---------------------------------------------------------------------------
// Kernel 1: swizzle W (f32 [C,H]) -> bf16 B-fragment layout
// layout: wfrag[w][kstep(32)][ntile(8)][lane(32)][8 dwords]
// B-frag (32x16 bf16): lane%16 = col N, lane/16 selects K half (0..15 / 16..31),
// dword i packs (K=2i, K=2i+1) low-first.
// ---------------------------------------------------------------------------
__global__ void __launch_bounds__(32) k_build_wfrag(const float* __restrict__ Wq,
    const float* __restrict__ Wk, const float* __restrict__ Wv,
    uint32_t* __restrict__ wfrag) {
  int id = blockIdx.x;                 // 3*32*8 blocks
  int nt = id & 7, ks = (id >> 3) & 31, w = id >> 8;
  const float* W = (w == 0) ? Wq : (w == 1) ? Wk : Wv;
  int lane = threadIdx.x;
  int col = nt * 16 + (lane & 15);
  int k0  = ks * 32 + (lane >> 4) * 16;
  uint32_t* dst = wfrag + ((((size_t)w * 32 + ks) * 8 + nt) * 32 + lane) * 8;
#pragma unroll
  for (int i = 0; i < 8; ++i) {
    float lo = W[(size_t)(k0 + 2 * i) * H_ + col];
    float hi = W[(size_t)(k0 + 2 * i + 1) * H_ + col];
    dst[i] = pack2_bf16(lo, hi);
  }
}

// ---------------------------------------------------------------------------
// Kernel 2: fused QKV projection. 256 thr (8 waves), each wave = 16 rows x H.
// Q -> row-major bf16 [B*T][H]; K,V -> pre-swizzled B-fragment layouts for attn.
// ---------------------------------------------------------------------------
__global__ void __launch_bounds__(256) k_proj_qkv(const float* __restrict__ x,
    const uint32_t* __restrict__ wfrag, uint16_t* __restrict__ qb,
    uint16_t* __restrict__ kfrag16, uint16_t* __restrict__ vfrag16) {
  const int tid  = threadIdx.x;
  const int lane = tid & 31, wv = tid >> 5;
  const int ln = lane & 15, half = lane >> 4;
  const int m0 = blockIdx.x * 128 + wv * 16;       // global row in [0, B*T)

  const v8f vzero = {0.f,0.f,0.f,0.f,0.f,0.f,0.f,0.f};
  v8f acc[3][8];
#pragma unroll
  for (int a = 0; a < 3; ++a)
#pragma unroll
    for (int n = 0; n < 8; ++n) acc[a][n] = vzero;

  const float* xr = x + (size_t)(m0 + ln) * C_;
  for (int ks = 0; ks < C_ / 32; ++ks) {
    // A-fragment from x: one contiguous 16-float run per lane, cvt to bf16
    Frag16 a;
    int k0 = ks * 32 + half * 16;
    float4 f0 = *(const float4*)(xr + k0);
    float4 f1 = *(const float4*)(xr + k0 + 4);
    float4 f2 = *(const float4*)(xr + k0 + 8);
    float4 f3 = *(const float4*)(xr + k0 + 12);
    a.u[0] = pack2_bf16(f0.x, f0.y); a.u[1] = pack2_bf16(f0.z, f0.w);
    a.u[2] = pack2_bf16(f1.x, f1.y); a.u[3] = pack2_bf16(f1.z, f1.w);
    a.u[4] = pack2_bf16(f2.x, f2.y); a.u[5] = pack2_bf16(f2.z, f2.w);
    a.u[6] = pack2_bf16(f3.x, f3.y); a.u[7] = pack2_bf16(f3.z, f3.w);
#pragma unroll
    for (int w3 = 0; w3 < 3; ++w3) {
#pragma unroll
      for (int nt = 0; nt < 8; ++nt) {
        Frag16 bf;
        const uint4* p = (const uint4*)(wfrag +
            ((((size_t)w3 * 32 + ks) * 8 + nt) * 32 + lane) * 8);
        bf.q[0] = p[0]; bf.q[1] = p[1];
        acc[w3][nt] = __builtin_amdgcn_wmma_f32_16x16x32_bf16(
            false, a.b, false, bf.b, (short)0, acc[w3][nt], false, false);
      }
    }
  }

  // ---- Q: row-major bf16 ----
#pragma unroll
  for (int nt = 0; nt < 8; ++nt) {
    int col = nt * 16 + ln;
#pragma unroll
    for (int r = 0; r < 8; ++r) {
      int grow = m0 + r + 8 * half;
      qb[(size_t)grow * H_ + col] = f2bf(acc[0][nt][r]);
    }
  }
  // ---- K: B-fragment layout kfrag[b][ktile(128)][hstep(4)][lane(32)][16 halves]
#pragma unroll
  for (int nt = 0; nt < 8; ++nt) {
    int h = nt * 16 + ln;
    int hstep = h >> 5, jj = h & 15, dw = jj >> 1, hw = jj & 1;
#pragma unroll
    for (int r = 0; r < 8; ++r) {
      int grow = m0 + r + 8 * half;
      int bb = grow >> 11, t = grow & (T_ - 1);
      int ktile = t >> 4;
      int lidx = (t & 15) + 16 * ((h & 31) >> 4);
      size_t idx = ((((size_t)bb * 128 + ktile) * 4 + hstep) * 32 + lidx) * 16 + dw * 2 + hw;
      kfrag16[idx] = f2bf(acc[1][nt][r]);
    }
  }
  // ---- V: B-fragment layout vfrag[b][kstep(64)][htile(8)][lane(32)][16 halves]
#pragma unroll
  for (int nt = 0; nt < 8; ++nt) {
    int h = nt * 16 + ln;
    int htile = h >> 4;
#pragma unroll
    for (int r = 0; r < 8; ++r) {
      int grow = m0 + r + 8 * half;
      int bb = grow >> 11, t = grow & (T_ - 1);
      int kst = t >> 5, jj = t & 15, dw = jj >> 1, hw = jj & 1;
      int lidx = (h & 15) + 16 * ((t & 31) >> 4);
      size_t idx = ((((size_t)bb * 64 + kst) * 8 + htile) * 32 + lidx) * 16 + dw * 2 + hw;
      vfrag16[idx] = f2bf(acc[2][nt][r]);
    }
  }
}

// ---------------------------------------------------------------------------
// Kernel 3: flash attention. Grid (T/128, B), 256 thr (8 waves).
// Wave = 16 query rows x H=128 output; 64-key blocks, online softmax.
// ---------------------------------------------------------------------------
__global__ void __launch_bounds__(256) k_attn(const uint16_t* __restrict__ qb,
    const uint32_t* __restrict__ kfrag, const uint32_t* __restrict__ vfrag,
    float* __restrict__ out) {
  __shared__ __align__(16) uint16_t pbuf[8][16][64];   // per-wave P staging, 16 KB
  const int tid  = threadIdx.x;
  const int lane = tid & 31, wv = tid >> 5;
  const int ln = lane & 15, half = lane >> 4;
  const int b  = blockIdx.y;
  const int m0 = blockIdx.x * 128 + wv * 16;           // query row within batch

  // Q A-fragments, one per 32-wide h step
  Frag16 qf[4];
  {
    const uint16_t* qrow = qb + ((size_t)b * T_ + (m0 + ln)) * H_;
#pragma unroll
    for (int hs = 0; hs < 4; ++hs) {
      qf[hs].q[0] = *(const uint4*)(qrow + hs * 32 + half * 16);
      qf[hs].q[1] = *(const uint4*)(qrow + hs * 32 + half * 16 + 8);
    }
  }

  const v8f vzero = {0.f,0.f,0.f,0.f,0.f,0.f,0.f,0.f};
  v8f o[8];
#pragma unroll
  for (int n = 0; n < 8; ++n) o[n] = vzero;
  float mi[8], li[8];
#pragma unroll
  for (int r = 0; r < 8; ++r) { mi[r] = -1e30f; li[r] = 0.f; }

  const float scale = 0.03125f;                        // C^-0.5 (reference uses C!)
  const int nb = (m0 + 15) / 64 + 1;                   // causal key-block count

  for (int j = 0; j < nb; ++j) {
    // ---- S = scale * Q K^T for 4 key subtiles ----
    v8f sb[4];
#pragma unroll
    for (int kt = 0; kt < 4; ++kt) {
      v8f s = vzero;
#pragma unroll
      for (int hs = 0; hs < 4; ++hs) {
        Frag16 kf;
        const uint4* p = (const uint4*)(kfrag +
            ((((size_t)b * 128 + (j * 4 + kt)) * 4 + hs) * 32 + lane) * 8);
        kf.q[0] = p[0]; kf.q[1] = p[1];
        s = __builtin_amdgcn_wmma_f32_16x16x32_bf16(
            false, qf[hs].b, false, kf.b, (short)0, s, false, false);
      }
      s = s * scale;
      int key = j * 64 + kt * 16 + ln;
#pragma unroll
      for (int r = 0; r < 8; ++r)
        if (key > m0 + r + 8 * half) s[r] = -1e30f;    // causal mask
      sb[kt] = s;
    }
    // ---- online softmax: row max across 4 subtiles + 16 lanes of the half ----
    float alpha[8];
#pragma unroll
    for (int r = 0; r < 8; ++r) {
      float mx = fmaxf(fmaxf(sb[0][r], sb[1][r]), fmaxf(sb[2][r], sb[3][r]));
#pragma unroll
      for (int msk = 1; msk < 16; msk <<= 1)
        mx = fmaxf(mx, __shfl_xor(mx, msk, 32));
      float mnew = fmaxf(mi[r], mx);
      alpha[r] = __expf(mi[r] - mnew);
      mi[r] = mnew;
    }
    float rs[8];
#pragma unroll
    for (int r = 0; r < 8; ++r) rs[r] = 0.f;
#pragma unroll
    for (int kt = 0; kt < 4; ++kt) {
#pragma unroll
      for (int r = 0; r < 8; ++r) {
        float p = __expf(sb[kt][r] - mi[r]);
        rs[r] += p;
        pbuf[wv][r + 8 * half][kt * 16 + ln] = f2bf(p); // C-layout -> [row][key]
      }
    }
#pragma unroll
    for (int r = 0; r < 8; ++r) {
#pragma unroll
      for (int msk = 1; msk < 16; msk <<= 1)
        rs[r] += __shfl_xor(rs[r], msk, 32);
      li[r] = li[r] * alpha[r] + rs[r];
#pragma unroll
      for (int n = 0; n < 8; ++n) o[n][r] *= alpha[r];
    }
    // ---- P (LDS, same-wave ordered) -> A-fragments; O += P V ----
    Frag16 pf[2];
#pragma unroll
    for (int k2 = 0; k2 < 2; ++k2) {
      pf[k2].q[0] = *(const uint4*)(&pbuf[wv][ln][k2 * 32 + half * 16]);
      pf[k2].q[1] = *(const uint4*)(&pbuf[wv][ln][k2 * 32 + half * 16 + 8]);
    }
#pragma unroll
    for (int nt = 0; nt < 8; ++nt) {
#pragma unroll
      for (int k2 = 0; k2 < 2; ++k2) {
        Frag16 vf;
        const uint4* p = (const uint4*)(vfrag +
            ((((size_t)b * 64 + (j * 2 + k2)) * 8 + nt) * 32 + lane) * 8);
        vf.q[0] = p[0]; vf.q[1] = p[1];
        o[nt] = __builtin_amdgcn_wmma_f32_16x16x32_bf16(
            false, pf[k2].b, false, vf.b, (short)0, o[nt], false, false);
      }
    }
  }

  // ---- epilogue: normalize and store fp32 ----
#pragma unroll
  for (int r = 0; r < 8; ++r) {
    float inv = 1.0f / li[r];
    int row = m0 + r + 8 * half;
#pragma unroll
    for (int nt = 0; nt < 8; ++nt)
      out[((size_t)b * T_ + row) * H_ + nt * 16 + ln] = o[nt][r] * inv;
  }
}

// ---------------------------------------------------------------------------
extern "C" void kernel_launch(void* const* d_in, const int* in_sizes, int n_in,
                              void* d_out, int out_size, void* d_ws, size_t ws_size,
                              hipStream_t stream) {
  (void)in_sizes; (void)n_in; (void)out_size; (void)ws_size;
  const float* x  = (const float*)d_in[0];
  const float* Wq = (const float*)d_in[1];
  const float* Wk = (const float*)d_in[2];
  const float* Wv = (const float*)d_in[3];
  float* out = (float*)d_out;
  uint8_t* ws = (uint8_t*)d_ws;

  const size_t WFRAG_BYTES = (size_t)3 * 32 * 8 * 32 * 8 * 4;    // 768 KB
  const size_t QB_BYTES    = (size_t)B_ * T_ * H_ * 2;           // 4 MB
  const size_t KF_BYTES    = (size_t)B_ * 128 * 4 * 32 * 8 * 4;  // 4 MB

  uint32_t* wfrag = (uint32_t*)(ws);
  uint16_t* qb    = (uint16_t*)(ws + WFRAG_BYTES);
  uint32_t* kfrag = (uint32_t*)(ws + WFRAG_BYTES + QB_BYTES);
  uint32_t* vfrag = (uint32_t*)(ws + WFRAG_BYTES + QB_BYTES + KF_BYTES);

  k_build_wfrag<<<3 * 32 * 8, 32, 0, stream>>>(Wq, Wk, Wv, wfrag);
  k_proj_qkv<<<(B_ * T_) / 128, 256, 0, stream>>>(x, wfrag, qb,
                                                  (uint16_t*)kfrag, (uint16_t*)vfrag);
  k_attn<<<dim3(T_ / 128, B_), 256, 0, stream>>>(qb, kfrag, vfrag, out);
}
